// TKGAT_9079560864592
// MI455X (gfx1250) — compile-verified, matrix operand order checked by reading, not run on hardware
//
#include <hip/hip_runtime.h>
#include <stdint.h>

typedef __attribute__((ext_vector_type(16))) __bf16 v16bf;
typedef __attribute__((ext_vector_type(8)))  float  v8f;

#define NN 100000      // nodes
#define NR 8           // relations
#define NE 1600000     // edges
#define ED_ 64
#define H_ 4
#define NU 1000
#define NUP 1008       // users padded to 16
#define NI 4096
#define CD 112         // concat dim 64+32+16
#define CDP 128        // padded K for WMMA

union ABu { v16bf v; uint4 u[2]; };

// ---------------- K1: W_ra^T (bf16) and r_emb_a ----------------
__global__ void k_prep(const float* __restrict__ W_R, const float* __restrict__ W_A,
                       const float* __restrict__ rel_embed,
                       __bf16* __restrict__ WraT, float* __restrict__ r_emb_a) {
  int r = blockIdx.x;
  for (int t = threadIdx.x; t < 64 * 64; t += blockDim.x) {
    int i = t >> 6, k2 = t & 63;
    const float* wr = W_R + ((long long)r * 64 + i) * 64;
    float acc = 0.f;
    #pragma unroll 8
    for (int j = 0; j < 64; ++j) acc += wr[j] * W_A[j * 64 + k2];
    WraT[((long long)r * 64 + k2) * 64 + i] = (__bf16)acc;   // transposed: [r][k2][i]
  }
  if (threadIdx.x < 64) {
    int k2 = threadIdx.x;
    float acc = 0.f;
    #pragma unroll 8
    for (int j = 0; j < 64; ++j) acc += rel_embed[r * 64 + j] * W_A[j * 64 + k2];
    r_emb_a[r * 64 + k2] = acc;
  }
}

// ---------------- K2: fp32 -> bf16 convert ----------------
__global__ void k_cvt(const float* __restrict__ x, __bf16* __restrict__ y, long long n) {
  long long i = (long long)blockIdx.x * blockDim.x + threadIdx.x;
  if (i < n) y[i] = (__bf16)x[i];
}

// ---------------- K3: proj[r] = E_bf16 @ W_ra[r]  (WMMA bf16) ----------------
__global__ void k_proj(const __bf16* __restrict__ Ebf, const __bf16* __restrict__ WraT,
                       float* __restrict__ proj) {
  const int r    = blockIdx.y;
  const int wave = threadIdx.x >> 5;
  const int lane = threadIdx.x & 31;
  const int tile = blockIdx.x * 8 + wave;          // node tile (16 rows)
  if (tile >= NN / 16) return;                     // wave-uniform exit
  const int half = (lane < 16) ? 0 : 8;            // K-offset for this lane half
  const int l15  = lane & 15;

  const __bf16* arow = Ebf + (long long)(tile * 16 + l15) * ED_;
  ABu a0, a1;
  a0.u[0] = *(const uint4*)(arow + half);          // K 0..7 / 8..15
  a0.u[1] = *(const uint4*)(arow + half + 16);     // K 16..23 / 24..31
  a1.u[0] = *(const uint4*)(arow + 32 + half);     // K 32..
  a1.u[1] = *(const uint4*)(arow + 32 + half + 16);

  const __bf16* wr = WraT + (long long)r * 64 * 64;
  float* pout = proj + ((long long)r * NN + tile * 16) * ED_;

  #pragma unroll
  for (int ct = 0; ct < 4; ++ct) {                 // 4 output column tiles of 16
    const __bf16* brow = wr + (long long)(ct * 16 + l15) * 64;   // column of B
    ABu b0, b1;
    b0.u[0] = *(const uint4*)(brow + half);
    b0.u[1] = *(const uint4*)(brow + half + 16);
    b1.u[0] = *(const uint4*)(brow + 32 + half);
    b1.u[1] = *(const uint4*)(brow + 32 + half + 16);
    v8f c = {};
    c = __builtin_amdgcn_wmma_f32_16x16x32_bf16(false, a0.v, false, b0.v, (short)0, c, false, false);
    c = __builtin_amdgcn_wmma_f32_16x16x32_bf16(false, a1.v, false, b1.v, (short)0, c, false, false);
    #pragma unroll
    for (int i = 0; i < 8; ++i)
      pout[(long long)(half + i) * ED_ + ct * 16 + l15] = c[i];
  }
}

// ---------------- K4: edge attention logits + segment max ----------------
__device__ __forceinline__ unsigned enc_f(float f) {
  unsigned b = __float_as_uint(f);
  return (b & 0x80000000u) ? ~b : (b | 0x80000000u);
}
__global__ void k_att(const int* __restrict__ src, const int* __restrict__ dst,
                      const int* __restrict__ etype, const float* __restrict__ proj,
                      const float* __restrict__ r_emb_a,
                      float* __restrict__ att, unsigned* __restrict__ m_enc) {
  __shared__ float s_rel[NR * 64];
  for (int t = threadIdx.x; t < NR * 64; t += blockDim.x) s_rel[t] = r_emb_a[t];
  __syncthreads();
  long long e = (long long)blockIdx.x * blockDim.x + threadIdx.x;
  if (e >= NE) return;
  int s = src[e], d = dst[e], r = etype[e];
  const float4* pt = (const float4*)(proj + ((long long)r * NN + s) * ED_);
  const float4* ph = (const float4*)(proj + ((long long)r * NN + d) * ED_);
  const float4* re = (const float4*)(s_rel + r * 64);
  float acc[4] = {0.f, 0.f, 0.f, 0.f};
  #pragma unroll 4
  for (int it = 0; it < 16; ++it) {
    float4 a = pt[it], b = ph[it], c = re[it];
    int h = it >> 2;
    acc[h] += tanhf(a.x) * tanhf(b.x + c.x);
    acc[h] += tanhf(a.y) * tanhf(b.y + c.y);
    acc[h] += tanhf(a.z) * tanhf(b.z + c.z);
    acc[h] += tanhf(a.w) * tanhf(b.w + c.w);
  }
  #pragma unroll
  for (int h = 0; h < 4; ++h) {
    att[e * 4 + h] = acc[h];
    atomicMax(&m_enc[d * 4 + h], enc_f(acc[h]));
  }
}

// ---------------- K5: e = exp(att - m), z += e ----------------
__global__ void k_exp(const int* __restrict__ dst, const unsigned* __restrict__ m_enc,
                      float* __restrict__ att, float* __restrict__ z) {
  long long e = (long long)blockIdx.x * blockDim.x + threadIdx.x;
  if (e >= NE) return;
  int d = dst[e];
  #pragma unroll
  for (int h = 0; h < 4; ++h) {
    unsigned enc = m_enc[d * 4 + h];
    unsigned bits = (enc & 0x80000000u) ? (enc ^ 0x80000000u) : ~enc;
    float m = __uint_as_float(bits);
    if (!isfinite(m)) m = 0.f;                    // matches jnp.where(isfinite, m, 0)
    float ev = expf(att[e * 4 + h] - m);
    att[e * 4 + h] = ev;
    unsafeAtomicAdd(&z[d * 4 + h], ev);
  }
}

// ---------------- K6: normalize ----------------
__global__ void k_norm(const int* __restrict__ dst, const float* __restrict__ z,
                       float* __restrict__ att) {
  long long e = (long long)blockIdx.x * blockDim.x + threadIdx.x;
  if (e >= NE) return;
  int d = dst[e];
  #pragma unroll
  for (int h = 0; h < 4; ++h)
    att[e * 4 + h] /= fmaxf(z[d * 4 + h], 1e-12f);
}

// ---------------- K7/K9: scatter-aggregate  N_h[d][h][:] += feat[s][:] * att[e][h] ----------------
template <int DIM>
__global__ void k_agg(const int* __restrict__ src, const int* __restrict__ dst,
                      const float* __restrict__ feat, const float* __restrict__ att,
                      float* __restrict__ Nh) {
  long long e = (long long)blockIdx.x * blockDim.x + threadIdx.x;
  if (e >= NE) return;
  int s = src[e], d = dst[e];
  float a[4];
  #pragma unroll
  for (int h = 0; h < 4; ++h) a[h] = att[e * 4 + h];
  const float* f = feat + (long long)s * DIM;
  float* out = Nh + (long long)d * 4 * DIM;
  for (int k = 0; k < DIM; k += 4) {
    float4 v = *(const float4*)(f + k);
    float vv[4] = {v.x, v.y, v.z, v.w};
    #pragma unroll
    for (int h = 0; h < 4; ++h)
      #pragma unroll
      for (int c = 0; c < 4; ++c)
        unsafeAtomicAdd(&out[h * DIM + k + c], vv[c] * a[h]);
  }
}

// ---------------- K8/K10: bi-interaction transform ----------------
template <int IN, int OUT>   // OUT = out_dim per head
__global__ void k_bi(const float* __restrict__ ego, const float* __restrict__ Nh,
                     const float* __restrict__ W1, const float* __restrict__ b1,
                     const float* __restrict__ W2, const float* __restrict__ b2,
                     float* __restrict__ out) {
  __shared__ float sW1[IN * OUT], sW2[IN * OUT], sb1[OUT], sb2[OUT];
  for (int t = threadIdx.x; t < IN * OUT; t += blockDim.x) { sW1[t] = W1[t]; sW2[t] = W2[t]; }
  if (threadIdx.x < OUT) { sb1[threadIdx.x] = b1[threadIdx.x]; sb2[threadIdx.x] = b2[threadIdx.x]; }
  __syncthreads();
  long long t = (long long)blockIdx.x * blockDim.x + threadIdx.x;
  if (t >= (long long)NN * H_) return;
  int n = (int)(t >> 2), h = (int)(t & 3);
  const float* eg = ego + (long long)n * IN;
  const float* nh = Nh + ((long long)n * 4 + h) * IN;
  float acc1[OUT], acc2[OUT];
  #pragma unroll
  for (int j = 0; j < OUT; ++j) { acc1[j] = 0.f; acc2[j] = 0.f; }
  for (int k = 0; k < IN; ++k) {
    float ev = eg[k], nv = nh[k];
    float x1 = ev + nv, x2 = ev * nv;
    #pragma unroll
    for (int j = 0; j < OUT; ++j) {
      acc1[j] += x1 * sW1[k * OUT + j];
      acc2[j] += x2 * sW2[k * OUT + j];
    }
  }
  float* o = out + (long long)n * (4 * OUT) + h * OUT;
  #pragma unroll
  for (int j = 0; j < OUT; ++j) {
    float y1 = acc1[j] + sb1[j]; y1 = (y1 > 0.f) ? y1 : 0.01f * y1;
    float y2 = acc2[j] + sb2[j]; y2 = (y2 > 0.f) ? y2 : 0.01f * y2;
    o[j] = y1 + y2;
  }
}

// ---------------- K11: gather rows, l2norm, concat -> bf16 (K padded to 128) ----------------
__global__ void k_build(const int* __restrict__ ids, int nids,
                        const float* __restrict__ ego, const float* __restrict__ ego1,
                        const float* __restrict__ ego2, __bf16* __restrict__ out) {
  int row = blockIdx.x;       // padded row count = gridDim.x
  int j = threadIdx.x;        // 128 threads
  float val = 0.f;
  if (row < nids) {
    int n = ids[row];
    if (j < 64) {
      val = ego[(long long)n * 64 + j];
    } else if (j < 96) {
      const float* p = ego1 + (long long)n * 32;
      float s = 0.f;
      #pragma unroll
      for (int k = 0; k < 32; ++k) s += p[k] * p[k];
      val = p[j - 64] / fmaxf(sqrtf(s), 1e-12f);
    } else if (j < CD) {
      const float* p = ego2 + (long long)n * 16;
      float s = 0.f;
      #pragma unroll
      for (int k = 0; k < 16; ++k) s += p[k] * p[k];
      val = p[j - 96] / fmaxf(sqrtf(s), 1e-12f);
    }
  }
  out[(long long)row * CDP + j] = (__bf16)val;
}

// ---------------- K12: out = U(1008x128) @ I(4096x128)^T  (WMMA bf16) ----------------
__global__ void k_out(const __bf16* __restrict__ U, const __bf16* __restrict__ I,
                      float* __restrict__ out) {
  const int wave = threadIdx.x >> 5;
  const int lane = threadIdx.x & 31;
  const int tid = blockIdx.x * 8 + wave;
  const int TM = NUP / 16, TN = NI / 16;
  if (tid >= TM * TN) return;
  const int tm = tid / TN, tn = tid % TN;
  const int half = (lane < 16) ? 0 : 8;
  const int l15 = lane & 15;
  const __bf16* arow = U + (long long)(tm * 16 + l15) * CDP;
  const __bf16* brow = I + (long long)(tn * 16 + l15) * CDP;   // I row = B column
  v8f c = {};
  #pragma unroll
  for (int kc = 0; kc < 4; ++kc) {
    ABu a, b;
    a.u[0] = *(const uint4*)(arow + kc * 32 + half);
    a.u[1] = *(const uint4*)(arow + kc * 32 + half + 16);
    b.u[0] = *(const uint4*)(brow + kc * 32 + half);
    b.u[1] = *(const uint4*)(brow + kc * 32 + half + 16);
    c = __builtin_amdgcn_wmma_f32_16x16x32_bf16(false, a.v, false, b.v, (short)0, c, false, false);
  }
  int col = tn * 16 + l15;
  int mbase = tm * 16 + half;
  #pragma unroll
  for (int i = 0; i < 8; ++i) {
    int row = mbase + i;
    if (row < NU) out[(long long)row * NI + col] = c[i];
  }
}

// ---------------- host ----------------
extern "C" void kernel_launch(void* const* d_in, const int* in_sizes, int n_in,
                              void* d_out, int out_size, void* d_ws, size_t ws_size,
                              hipStream_t stream) {
  const int*   src   = (const int*)d_in[0];
  const int*   dst   = (const int*)d_in[1];
  const int*   etyp  = (const int*)d_in[2];
  // d_in[3] = node_ids (arange -> identity gather, unused)
  const int*   uids  = (const int*)d_in[4];
  const int*   iids  = (const int*)d_in[5];
  const float* E     = (const float*)d_in[6];
  const float* relE  = (const float*)d_in[7];
  const float* W_R   = (const float*)d_in[8];
  const float* W_A   = (const float*)d_in[9];
  const float* W1_0  = (const float*)d_in[10];
  const float* b1_0  = (const float*)d_in[11];
  const float* W2_0  = (const float*)d_in[12];
  const float* b2_0  = (const float*)d_in[13];
  const float* W1_1  = (const float*)d_in[14];
  const float* b1_1  = (const float*)d_in[15];
  const float* W2_1  = (const float*)d_in[16];
  const float* b2_1  = (const float*)d_in[17];
  float* out = (float*)d_out;

  char* ws = (char*)d_ws;
  size_t off = 0;
  auto carve = [&](size_t bytes) -> void* {
    void* p = ws + off;
    off += (bytes + 255) & ~(size_t)255;
    return p;
  };
  __bf16*   WraT   = (__bf16*)  carve((size_t)NR * 64 * 64 * 2);
  float*    rEmbA  = (float*)   carve((size_t)NR * 64 * 4);
  __bf16*   Ebf    = (__bf16*)  carve((size_t)NN * ED_ * 2);
  float*    proj   = (float*)   carve((size_t)NR * NN * ED_ * 4);
  float*    att    = (float*)   carve((size_t)NE * 4 * 4);
  unsigned* m_enc  = (unsigned*)carve((size_t)NN * 4 * 4);
  float*    z      = (float*)   carve((size_t)NN * 4 * 4);
  float*    Nh0    = (float*)   carve((size_t)NN * 4 * 64 * 4);
  float*    ego1   = (float*)   carve((size_t)NN * 32 * 4);
  float*    Nh1    = (float*)   carve((size_t)NN * 4 * 32 * 4);
  float*    ego2   = (float*)   carve((size_t)NN * 16 * 4);
  __bf16*   Ubf    = (__bf16*)  carve((size_t)NUP * CDP * 2);
  __bf16*   Ibf    = (__bf16*)  carve((size_t)NI * CDP * 2);

  // zero the atomic accumulators (graph-capture-safe, deterministic per call)
  hipMemsetAsync(m_enc, 0, (size_t)NN * 4 * 4, stream);          // 0 == minimum of float encoding
  hipMemsetAsync(z,     0, (size_t)NN * 4 * 4, stream);
  hipMemsetAsync(Nh0,   0, (size_t)NN * 4 * 64 * 4, stream);
  hipMemsetAsync(Nh1,   0, (size_t)NN * 4 * 32 * 4, stream);

  k_prep<<<NR, 256, 0, stream>>>(W_R, W_A, relE, WraT, rEmbA);
  k_cvt<<<(NN * ED_ + 255) / 256, 256, 0, stream>>>(E, Ebf, (long long)NN * ED_);
  k_proj<<<dim3((NN / 16 + 7) / 8, NR), 256, 0, stream>>>(Ebf, WraT, proj);
  k_att<<<NE / 256, 256, 0, stream>>>(src, dst, etyp, proj, rEmbA, att, m_enc);
  k_exp<<<NE / 256, 256, 0, stream>>>(dst, m_enc, att, z);
  k_norm<<<NE / 256, 256, 0, stream>>>(dst, z, att);
  k_agg<64><<<NE / 256, 256, 0, stream>>>(src, dst, E, att, Nh0);
  k_bi<64, 8><<<(NN * H_ + 255) / 256, 256, 0, stream>>>(E, Nh0, W1_0, b1_0, W2_0, b2_0, ego1);
  k_agg<32><<<NE / 256, 256, 0, stream>>>(src, dst, ego1, att, Nh1);
  k_bi<32, 4><<<(NN * H_ + 255) / 256, 256, 0, stream>>>(ego1, Nh1, W1_1, b1_1, W2_1, b2_1, ego2);
  k_build<<<NUP, CDP, 0, stream>>>(uids, NU, E, ego1, ego2, Ubf);
  k_build<<<NI, CDP, 0, stream>>>(iids, NI, E, ego1, ego2, Ibf);
  k_out<<<((NUP / 16) * (NI / 16) + 7) / 8, 256, 0, stream>>>(Ubf, Ibf, out);
}
